// DeformableAttnBlock_87445534147210
// MI455X (gfx1250) — compile-verified, matrix operand order checked by reading, not run on hardware
//
#include <hip/hip_runtime.h>
#include <hip/hip_bf16.h>

// ---- problem constants (match reference) ----
#define BB 2
#define CC 64
#define KK 25
#define HH 160
#define WW 160
#define HW (HH * WW)          // 25600
#define NPIX (BB * HW)        // 51200
#define NOUT 75               // 3*K param channels
#define PCH 80                // padded to 5 WMMA n-tiles
#define MAXOFF 4.0f

typedef _Float16 v16h __attribute__((ext_vector_type(16)));
typedef _Float16 v8h  __attribute__((ext_vector_type(8)));
typedef float    v8f  __attribute__((ext_vector_type(8)));
typedef float    v4f  __attribute__((ext_vector_type(4)));

static __device__ __forceinline__ v8f wmma_f16(v16h a, v16h b, v8f c) {
    // D = A(16x32 f16) x B(32x16 f16) + C(16x16 f32)
    return __builtin_amdgcn_wmma_f32_16x16x32_f16(
        /*neg_a=*/false, a, /*neg_b=*/false, b,
        /*c_mod=*/(short)0, c, /*reuse_a=*/false, /*reuse_b=*/false);
}

static __device__ __forceinline__ int reflect_idx(int i, int n) {
    // pad=1 reflect (no edge repeat): -1 -> 1, n -> n-2
    if (i < 0) i = -i;
    if (i >= n) i = 2 * n - 2 - i;
    return i;
}

static __device__ __forceinline__ float reflectf(float c, float mx) {
    float m = 2.0f * mx;
    c = fmodf(fabsf(c), m);
    return c > mx ? m - c : c;
}

// ============================================================
// Stage 1: depthwise 3x3 reflect conv + bias + exact GELU
// writes h as f16 in pixel-major layout: h16[(b*HW + p)*64 + c]
// ============================================================
__global__ __launch_bounds__(256)
void dwconv_gelu_kernel(const float* __restrict__ x,
                        const float* __restrict__ dww,
                        const float* __restrict__ dwb,
                        _Float16* __restrict__ h16) {
    int g  = blockIdx.x * 256 + threadIdx.x;   // (b, c, p) with p fastest
    int p  = g % HW;
    int bc = g / HW;                            // b*CC + c
    int c  = bc % CC;
    int b  = bc / CC;
    int xc = p % WW, yc = p / WW;

    const float* xp = x + (size_t)bc * HW;
    float acc = dwb[c];
#pragma unroll
    for (int ky = 0; ky < 3; ++ky) {
        int iy = reflect_idx(yc + ky - 1, HH);
#pragma unroll
        for (int kx = 0; kx < 3; ++kx) {
            int ix = reflect_idx(xc + kx - 1, WW);
            acc += xp[iy * WW + ix] * dww[c * 9 + ky * 3 + kx];
        }
    }
    // exact GELU: 0.5*x*(1+erf(x/sqrt(2)))
    float gelu = 0.5f * acc * (1.0f + erff(acc * 0.70710678118654752f));
    h16[(size_t)(b * HW + p) * CC + c] = (_Float16)gelu;
}

// ============================================================
// Stage 2: pointwise 64->75 GEMM via WMMA f32_16x16x32_f16,
// then bias + tanh offsets + softmax attention + reflect coords.
// Per wave: 16 pixels x 80 (padded) outputs = 5 n-tiles x 2 k-frags.
// smp[pix*75 + {0..24:ix, 25..49:iy, 50..74:attn}]
// ============================================================
__global__ __launch_bounds__(256)
void pconv_wmma_kernel(const _Float16* __restrict__ h16,
                       const float* __restrict__ pww,   // [75][64]
                       const float* __restrict__ pwb,   // [75]
                       const float* __restrict__ bofs,  // [25][2] (ox,oy)
                       float* __restrict__ smp) {
    __shared__ float lds[8][16][PCH];                   // 40 KB

    const int lane     = threadIdx.x & 31;
    const int w        = threadIdx.x >> 5;
    const int tileBase = (blockIdx.x * 8 + w) * 16;
    const int nOff     = lane & 15;                     // M (A) / N (B) within tile
    const int kbase    = (lane & 16) ? 8 : 0;           // K interleave per ISA layout

    // ---- A fragments: 16 pixels x 64 (two 16x32 f16 frags) ----
    const _Float16* hrow = h16 + (size_t)(tileBase + nOff) * CC;
    v16h a0, a1;
    {
        v8h lo = *(const v8h*)(hrow + kbase);           // K = kbase..kbase+7
        v8h hi = *(const v8h*)(hrow + kbase + 16);      // K = kbase+16..+23
#pragma unroll
        for (int i = 0; i < 8; ++i) { a0[i] = lo[i]; a0[8 + i] = hi[i]; }
        lo = *(const v8h*)(hrow + 32 + kbase);
        hi = *(const v8h*)(hrow + 32 + kbase + 16);
#pragma unroll
        for (int i = 0; i < 8; ++i) { a1[i] = lo[i]; a1[8 + i] = hi[i]; }
    }

    // ---- B fragments: W[o][k], 5 n-tiles x 2 k-frags (mirror of A layout) ----
    v16h bf[5][2];
#pragma unroll
    for (int nt = 0; nt < 5; ++nt) {
        int  o     = nt * 16 + nOff;
        bool valid = (o < NOUT);
        const float* wr = pww + (size_t)(valid ? o : 0) * CC;
#pragma unroll
        for (int kf = 0; kf < 2; ++kf) {
            v4f f0 = *(const v4f*)(wr + kf * 32 + kbase);
            v4f f1 = *(const v4f*)(wr + kf * 32 + kbase + 4);
            v4f f2 = *(const v4f*)(wr + kf * 32 + kbase + 16);
            v4f f3 = *(const v4f*)(wr + kf * 32 + kbase + 20);
            v16h t;
#pragma unroll
            for (int i = 0; i < 4; ++i) {
                t[i]      = (_Float16)(valid ? f0[i] : 0.0f);
                t[4 + i]  = (_Float16)(valid ? f1[i] : 0.0f);
                t[8 + i]  = (_Float16)(valid ? f2[i] : 0.0f);
                t[12 + i] = (_Float16)(valid ? f3[i] : 0.0f);
            }
            bf[nt][kf] = t;
        }
    }

    // ---- 10 WMMAs: accumulate 16x80 tile in f32 ----
    v8f acc[5] = {};
#pragma unroll
    for (int nt = 0; nt < 5; ++nt) {
        acc[nt] = wmma_f16(a0, bf[nt][0], acc[nt]);
        acc[nt] = wmma_f16(a1, bf[nt][1], acc[nt]);
    }

    // ---- D layout: lane%16 = N, VGPR r -> M = r + (lane>=16)*8 ----
    const int mHalf = (lane >> 4) * 8;
#pragma unroll
    for (int nt = 0; nt < 5; ++nt) {
        int   o    = nt * 16 + nOff;
        float bias = (o < NOUT) ? pwb[o] : 0.0f;
#pragma unroll
        for (int r = 0; r < 8; ++r)
            lds[w][mHalf + r][o] = acc[nt][r] + bias;
    }
    __syncthreads();

    // ---- per-pixel post-processing (lanes 0..15, one pixel each) ----
    if (lane < 16) {
        const int   pix = tileBase + lane;
        const float* L  = &lds[w][lane][0];
        const int p  = pix % HW;
        const int xc = p % WW, yc = p / WW;

        // softmax over 25 attention logits
        float mx = -1e30f;
#pragma unroll
        for (int k = 0; k < KK; ++k) mx = fmaxf(mx, L[2 * KK + k]);
        float e[KK], s = 0.0f;
#pragma unroll
        for (int k = 0; k < KK; ++k) { e[k] = __expf(L[2 * KK + k] - mx); s += e[k]; }
        const float inv = 1.0f / s;

        float* outp = smp + (size_t)pix * NOUT;
#pragma unroll
        for (int k = 0; k < KK; ++k) {
            float dx = bofs[2 * k]     + MAXOFF * tanhf(L[2 * k]);
            float dy = bofs[2 * k + 1] + MAXOFF * tanhf(L[2 * k + 1]);
            outp[k]          = reflectf((float)xc + dx, (float)(WW - 1));
            outp[KK + k]     = reflectf((float)yc + dy, (float)(HH - 1));
            outp[2 * KK + k] = e[k] * inv;
        }
    }
}

// ============================================================
// Stage 3: deformable bilinear sampling, attention-weighted sum.
// One 64-thread block per pixel; thread = channel.
// ============================================================
__global__ __launch_bounds__(64)
void sample_kernel(const float* __restrict__ x,
                   const float* __restrict__ smp,
                   float* __restrict__ out) {
    __shared__ float sp[NOUT];
    const int pix = blockIdx.x;
    const int c   = threadIdx.x;
    for (int i = c; i < NOUT; i += 64) sp[i] = smp[(size_t)pix * NOUT + i];
    __syncthreads();

    const int b = pix / HW;
    const int p = pix % HW;
    const float* xb = x + (size_t)(b * CC + c) * HW;

    float acc = 0.0f;
#pragma unroll
    for (int k = 0; k < KK; ++k) {
        float ix = sp[k], iy = sp[KK + k], at = sp[2 * KK + k];
        float x0f = floorf(ix), y0f = floorf(iy);
        float wx1 = ix - x0f, wy1 = iy - y0f;
        float wx0 = 1.0f - wx1, wy0 = 1.0f - wy1;
        int x0 = min(max((int)x0f, 0), WW - 1);
        int x1 = min((int)x0f + 1, WW - 1);
        int y0 = min(max((int)y0f, 0), HH - 1);
        int y1 = min((int)y0f + 1, HH - 1);
        float v00 = xb[y0 * WW + x0];
        float v01 = xb[y0 * WW + x1];
        float v10 = xb[y1 * WW + x0];
        float v11 = xb[y1 * WW + x1];
        acc += at * (v00 * wy0 * wx0 + v01 * wy0 * wx1 +
                     v10 * wy1 * wx0 + v11 * wy1 * wx1);
    }
    out[(size_t)(b * CC + c) * HW + p] = acc;
}

// ============================================================
extern "C" void kernel_launch(void* const* d_in, const int* in_sizes, int n_in,
                              void* d_out, int out_size, void* d_ws, size_t ws_size,
                              hipStream_t stream) {
    const float* x    = (const float*)d_in[0];
    const float* dww  = (const float*)d_in[1];
    const float* dwb  = (const float*)d_in[2];
    const float* pww  = (const float*)d_in[3];
    const float* pwb  = (const float*)d_in[4];
    const float* bofs = (const float*)d_in[5];

    _Float16* h16 = (_Float16*)d_ws;                               // 6.55 MB
    float*    smp = (float*)((char*)d_ws + (size_t)NPIX * CC * 2); // 15.36 MB

    // Stage 1: 3,276,800 elements / 256
    dwconv_gelu_kernel<<<(BB * CC * HW) / 256, 256, 0, stream>>>(x, dww, dwb, h16);
    // Stage 2: 3200 pixel tiles, 8 waves/block -> 400 blocks
    pconv_wmma_kernel<<<NPIX / (8 * 16), 256, 0, stream>>>(h16, pww, pwb, bofs, smp);
    // Stage 3: one block per pixel
    sample_kernel<<<NPIX, 64, 0, stream>>>(x, smp, (float*)d_out);
}
